// MPNN_block_seperate_49795850830070
// MI455X (gfx1250) — matrix operand order repaired and checked by reading, not compiled, not run on hardware
//
#include <hip/hip_runtime.h>
#include <math.h>

// ---------------------------------------------------------------------------
// Types / WMMA helpers (CDNA5 wave32, v_wmma_f32_16x16x32_bf16)
// ---------------------------------------------------------------------------
typedef __attribute__((ext_vector_type(16))) __bf16 v16bf;
typedef __attribute__((ext_vector_type(8)))  __bf16 bf16x8;
typedef __attribute__((ext_vector_type(8)))  float  v8f;
typedef __attribute__((ext_vector_type(4)))  unsigned int v4u;
typedef __attribute__((ext_vector_type(8)))  int v8i;
typedef __attribute__((ext_vector_type(4)))  int v4i;

#define DEVI static __device__ __forceinline__

DEVI v16bf mpnn_ld2(const __bf16* p0, const __bf16* p1) {
  bf16x8 lo = *(const bf16x8*)p0;
  bf16x8 hi = *(const bf16x8*)p1;
  v16bf r;
#pragma unroll
  for (int i = 0; i < 8; ++i) { r[i] = lo[i]; r[i + 8] = hi[i]; }
  return r;
}
// A fragment (16x32, M=lane&15): elems 0..7 -> K = kb+8g+e ; 8..15 -> K = kb+16+8g+e
DEVI v16bf mpnn_afrag(const __bf16* row, int kb, int g) {
  return mpnn_ld2(row + kb + g * 8, row + kb + g * 8 + 16);
}
// B fragment (32x16, N=lane&15): elems 0..7 -> K = kb+8g+e ; 8..15 -> K = kb+16+8g+e
DEVI v16bf mpnn_bfrag(const __bf16* col, int kb, int g) {
  return mpnn_ld2(col + kb + g * 8, col + kb + g * 8 + 16);
}
DEVI v8f mpnn_wmma(v16bf a, v16bf b, v8f c) {
  return __builtin_amdgcn_wmma_f32_16x16x32_bf16(false, a, false, b, (short)0, c,
                                                 false, false);
}
DEVI v8f mpnn_zero8() {
  v8f z;
#pragma unroll
  for (int i = 0; i < 8; ++i) z[i] = 0.0f;
  return z;
}
DEVI float mpnn_lrelu(float x) { return x > 0.0f ? x : 0.01f * x; }

// ---------------------------------------------------------------------------
// Tensor Data Mover: 2D tile [rows, cols] (bf16) global -> LDS offset 0.
// D# built per CDNA5 ISA ch.8: group0 = {count/flags, lds_addr, global_addr,
// type=2}; group1 = {data_size=2B, tensor_dim0/1, tile_dim0/1, dim0 stride}.
// Issued wave-uniform; completion via s_wait_tensorcnt.
// ---------------------------------------------------------------------------
DEVI void mpnn_tdm_load_2d(const void* gptr, unsigned lds_off, unsigned valid_w,
                           unsigned rows, unsigned tile_w, unsigned stride) {
  unsigned long long ga = (unsigned long long)(__SIZE_TYPE__)gptr;
  v4u g0;
  g0[0] = 1u;                                   // count=1, user descriptor
  g0[1] = lds_off;                              // lds_addr (bytes)
  g0[2] = (unsigned)(ga & 0xFFFFFFFFull);       // global_addr[31:0]
  g0[3] = (unsigned)((ga >> 32) & 0x1FFFFFFull) // global_addr[56:32]
          | (2u << 30);                         // type = 2 ("image")
  v8i g1;
  g1[0] = (int)(1u << 16);                      // data_size = 1 (2 bytes)
  g1[1] = (int)((valid_w & 0xFFFFu) << 16);     // tensor_dim0[15:0]
  g1[2] = (int)(((valid_w >> 16) & 0xFFFFu) | ((rows & 0xFFFFu) << 16));
  g1[3] = (int)(((rows >> 16) & 0xFFFFu) | ((tile_w & 0xFFFFu) << 16));
  g1[4] = (int)(rows & 0xFFFFu);                // tile_dim1 (tile_dim2 = 0)
  g1[5] = (int)stride;                          // tensor_dim0_stride[31:0]
  g1[6] = 0;                                    // stride[47:32], dim1_stride lo
  g1[7] = 0;
  v4i z4;
  z4[0] = 0; z4[1] = 0; z4[2] = 0; z4[3] = 0;
#if defined(__clang_major__) && (__clang_major__ >= 23)
  v8i z8;
#pragma unroll
  for (int i = 0; i < 8; ++i) z8[i] = 0;
  __builtin_amdgcn_tensor_load_to_lds(g0, g1, z4, z4, z8, 0);
#else
  __builtin_amdgcn_tensor_load_to_lds(g0, g1, z4, z4, 0);
#endif
  __builtin_amdgcn_s_wait_tensorcnt(0);
}

#define EPS_BN 1e-5f

// ---------------------------------------------------------------------------
// Utility kernels
// ---------------------------------------------------------------------------
__global__ void mpnn_zero_f32(float* p, int n) {
  int i = blockIdx.x * 256 + threadIdx.x;
  if (i < n) p[i] = 0.0f;
}

__global__ void mpnn_cvt_bf16(const float* __restrict__ in, __bf16* __restrict__ out,
                              int n) {
  int i = blockIdx.x * 256 + threadIdx.x;
  if (i < n) out[i] = (__bf16)in[i];
}

// x [16,256,64,64] f32 -> X0 [1024(b*n), 64(d), 256(s)] bf16
__global__ void mpnn_pack_x(const float* __restrict__ x, __bf16* __restrict__ X0) {
  int idx = blockIdx.x * 256 + threadIdx.x;  // total 16*256*64*64 = 16777216
  if (idx >= 16 * 256 * 64 * 64) return;
  int d = idx & 63;
  int n = (idx >> 6) & 63;
  int s = (idx >> 12) & 255;
  int b = idx >> 20;
  X0[(((size_t)(b * 64 + n)) * 64 + d) * 256 + s] = (__bf16)x[idx];
}

// ---------------------------------------------------------------------------
// Implicit-GEMM conv (K=3) with fused BN-stat accumulation.
// grid = (1024, COUT/64, ceil(LOUT/64)), block = 128 (4 waves).
// Input tile [CIN, 66] staged in LDS via the Tensor Data Mover for interior
// tiles (TDM zero-fills the right edge via tensor_dim0); the left-padded
// l0==0 tile uses a bounds-checked scalar path.
// Dynamic LDS layout: Xl @0 (CIN*132 B), Bt @CIN*132 (4096 B), sst (+512 B).
// ---------------------------------------------------------------------------
template <int CIN, int COUT, int LIN, int LOUT, int PAD>
__global__ void __launch_bounds__(128) mpnn_conv_bn(const __bf16* __restrict__ X,
                                                    const __bf16* __restrict__ W,
                                                    __bf16* __restrict__ Y,
                                                    float* __restrict__ stats) {
  constexpr int KDIM = CIN * 3;
  constexpr int XW = 66;  // 64 + (K-1)
  extern __shared__ char mpnn_smem[];
  __bf16* Xl = (__bf16*)mpnn_smem;
  __bf16* Bt = (__bf16*)(mpnn_smem + CIN * 132);
  float* sst = (float*)(mpnn_smem + CIN * 132 + 4096);

  const int b = blockIdx.x;
  const int ct = blockIdx.y;
  const int l0 = blockIdx.z * 64;
  const int tid = threadIdx.x;
  const int lane = tid & 31;
  const int wv = tid >> 5;
  const int g = lane >> 4;
  const int ln = lane & 15;

  sst[tid] = 0.0f;
  const __bf16* Xb = X + (size_t)b * CIN * LIN;
  if (l0 > 0) {
    // interior tile: async 2D DMA, rows = CIN, cols = 66 starting at l0-PAD
    if (wv == 0)
      mpnn_tdm_load_2d(Xb + (l0 - PAD), 0u, (unsigned)(LIN - (l0 - PAD)),
                       (unsigned)CIN, (unsigned)XW, (unsigned)LIN);
  } else {
    for (int idx = tid; idx < CIN * XW; idx += 128) {
      int ci = idx / XW, j = idx - ci * XW;
      int li = l0 - PAD + j;
      Xl[idx] = (li >= 0 && li < LIN) ? Xb[(size_t)ci * LIN + li] : (__bf16)0.0f;
    }
  }
  __syncthreads();

  v8f acc[4];
#pragma unroll
  for (int t = 0; t < 4; ++t) acc[t] = mpnn_zero8();

  const __bf16* Wrow = W + (size_t)(ct * 64 + wv * 16 + ln) * KDIM;

  for (int ks = 0; ks < KDIM / 32; ++ks) {
    // build 32-wide im2col slab: Bt[n][j], K = ks*32+j
    for (int idx = tid; idx < 64 * 32; idx += 128) {
      int n = idx >> 5, j = idx & 31;
      int K = ks * 32 + j;
      int ci = K / 3;
      int kk = K - ci * 3;
      Bt[idx] = Xl[ci * XW + n + kk];
    }
    __syncthreads();
    if (ks + 1 < KDIM / 32) __builtin_prefetch(Wrow + (ks + 1) * 32, 0, 1);
    v16bf af = mpnn_afrag(Wrow, ks * 32, g);
#pragma unroll
    for (int t = 0; t < 4; ++t) {
      v16bf bf_ = mpnn_bfrag(&Bt[(t * 16 + ln) * 32], 0, g);
      acc[t] = mpnn_wmma(af, bf_, acc[t]);
    }
    __syncthreads();
  }

  // epilogue: store bf16 conv output, accumulate per-channel sum / sumsq
#pragma unroll
  for (int t = 0; t < 4; ++t) {
    int n = l0 + t * 16 + ln;
    if (n < LOUT) {
#pragma unroll
      for (int i = 0; i < 8; ++i) {
        int ml = wv * 16 + g * 8 + i;  // local cout
        int m = ct * 64 + ml;
        float v = acc[t][i];
        Y[((size_t)b * COUT + m) * LOUT + n] = (__bf16)v;
        atomicAdd(&sst[ml], v);
        atomicAdd(&sst[64 + ml], v * v);
      }
    }
  }
  __syncthreads();
  int ch = ct * 64 + (tid & 63);
  float* dst = (tid < 64) ? &stats[ch] : &stats[COUT + ch];
  atomicAdd(dst, sst[tid]);
}

// stats -> per-channel scale/shift
__global__ void mpnn_finalize_bn(const float* __restrict__ st,
                                 const float* __restrict__ gam,
                                 const float* __restrict__ bet, float cnt, int C,
                                 float* __restrict__ ss) {
  int c = blockIdx.x * blockDim.x + threadIdx.x;
  if (c >= C) return;
  float mean = st[c] / cnt;
  float var = st[C + c] / cnt - mean * mean;
  float sc = gam[c] * rsqrtf(var + EPS_BN);
  ss[c] = sc;
  ss[C + c] = bet[c] - mean * sc;
}

// BN + ReLU + MaxPool1d(k=2,s=2,pad=1): window p covers inputs {2p-1, 2p}
template <int C, int LOUT, int LPOOL>
__global__ void mpnn_bn_relu_pool(const __bf16* __restrict__ in,
                                  const float* __restrict__ ss,
                                  __bf16* __restrict__ out) {
  long idx = (long)blockIdx.x * 256 + threadIdx.x;
  if (idx >= (long)1024 * C * LPOOL) return;
  int p = (int)(idx % LPOOL);
  long r = idx / LPOOL;
  int c = (int)(r % C);
  int b = (int)(r / C);
  float sc = ss[c], sh = ss[C + c];
  const __bf16* row = in + ((size_t)b * C + c) * LOUT;
  float m = -1e30f;
  int il = 2 * p - 1, ir = 2 * p;
  if (il >= 0 && il < LOUT) m = fmaxf(m, fmaxf((float)row[il] * sc + sh, 0.0f));
  if (ir >= 0 && ir < LOUT) m = fmaxf(m, fmaxf((float)row[ir] * sc + sh, 0.0f));
  out[idx] = (__bf16)m;
}

// t3 [1024, 256, 36] -> spa [1152(b*tc), 64(n), 128(f)]
__global__ void mpnn_pack_spa(const __bf16* __restrict__ t3,
                              __bf16* __restrict__ spa) {
  int idx = blockIdx.x * 256 + threadIdx.x;  // 1152*64*128 = 9437184
  if (idx >= 1152 * 64 * 128) return;
  int f = idx & 127;
  int n = (idx >> 7) & 63;
  int bt = idx >> 13;
  int b = bt / 72;
  int tc = bt - b * 72;
  int q = tc * 128 + f;  // flat over (c, l) of the [256,36] block
  int c = q / 36;
  int l = q - c * 36;
  spa[idx] = t3[(((size_t)(b * 64 + n)) * 256 + c) * 36 + l];
}

// ---------------------------------------------------------------------------
// nf = spa @ map_w^T + map_b : 1152 batches, [64,128] = [64,128]@[128,128]
// grid = 1152, block = 256 (8 waves; wave = one 16-wide output-col strip)
// ---------------------------------------------------------------------------
__global__ void __launch_bounds__(256) mpnn_nf_gemm(const __bf16* __restrict__ spa,
                                                    const __bf16* __restrict__ mw,
                                                    const float* __restrict__ mb,
                                                    __bf16* __restrict__ nf) {
  const int bt = blockIdx.x;
  const int tid = threadIdx.x;
  const int lane = tid & 31;
  const int wv = tid >> 5;
  const int g = lane >> 4;
  const int ln = lane & 15;

  v8f acc[4];
#pragma unroll
  for (int j = 0; j < 4; ++j) acc[j] = mpnn_zero8();

  const int oc = wv * 16 + ln;
  const __bf16* brow = mw + (size_t)oc * 128;  // map_w[o][f] : contiguous K
  const __bf16* abase = spa + (size_t)bt * 64 * 128;

#pragma unroll
  for (int ks = 0; ks < 4; ++ks) {
    v16bf bf_ = mpnn_bfrag(brow, ks * 32, g);
#pragma unroll
    for (int j = 0; j < 4; ++j) {
      v16bf af = mpnn_afrag(abase + (size_t)(j * 16 + ln) * 128, ks * 32, g);
      acc[j] = mpnn_wmma(af, bf_, acc[j]);
    }
  }
  float bias = mb[oc];
#pragma unroll
  for (int j = 0; j < 4; ++j)
#pragma unroll
    for (int i = 0; i < 8; ++i) {
      int m = j * 16 + g * 8 + i;
      nf[((size_t)bt * 64 + m) * 128 + oc] = (__bf16)(acc[j][i] + bias);
    }
}

// ---------------------------------------------------------------------------
// Fused graph kernel per batch bt:
//   adj = lrelu(nf nf^T - 1e8 I); adj = softmax(adj) + I
//   h = adj @ nf ; out = h @ theta^T + theta_b  (+ BN stat accumulation)
// grid = 1152, block = 128 (4 waves; wave = one 16-row M strip)
// ubuf holds adjF (f32, adjacency) first, then is reused as nfT (bf16,
// transposed nf) so the h-GEMM B fragments are contiguous ds_load_b128s.
// ---------------------------------------------------------------------------
__global__ void __launch_bounds__(128) mpnn_graph(const __bf16* __restrict__ nf,
                                                  const __bf16* __restrict__ thw,
                                                  const float* __restrict__ thb,
                                                  float* __restrict__ G,
                                                  float* __restrict__ gstats) {
  __shared__ alignas(16) __bf16 nfL[64 * 128];
  __shared__ alignas(16) char ubuf[64 * 64 * 4];  // adjF f32 | nfT bf16[128][64]
  __shared__ alignas(16) __bf16 adjB[64 * 64];
  __shared__ alignas(16) __bf16 hB[64 * 128];
  __shared__ float sst[128];
  float* adjF = (float*)ubuf;
  __bf16* nfT = (__bf16*)ubuf;

  const int bt = blockIdx.x;
  const int tid = threadIdx.x;
  const int lane = tid & 31;
  const int wv = tid >> 5;
  const int g = lane >> 4;
  const int ln = lane & 15;

  sst[tid] = 0.0f;
  const __bf16* src = nf + (size_t)bt * 64 * 128;
  for (int i = tid; i < 1024; i += 128)  // 8192 halves as bf16x8 chunks
    ((bf16x8*)nfL)[i] = ((const bf16x8*)src)[i];
  __syncthreads();

  // ---- adj = nf @ nf^T  (M=64,N=64,K=128) ----
  v8f acc[4];
#pragma unroll
  for (int t = 0; t < 4; ++t) acc[t] = mpnn_zero8();
#pragma unroll
  for (int ks = 0; ks < 4; ++ks) {
    v16bf af = mpnn_afrag(nfL + (size_t)(wv * 16 + ln) * 128, ks * 32, g);
#pragma unroll
    for (int t = 0; t < 4; ++t) {
      v16bf bf_ = mpnn_bfrag(nfL + (size_t)(t * 16 + ln) * 128, ks * 32, g);
      acc[t] = mpnn_wmma(af, bf_, acc[t]);
    }
  }
#pragma unroll
  for (int t = 0; t < 4; ++t)
#pragma unroll
    for (int i = 0; i < 8; ++i) {
      int m = wv * 16 + g * 8 + i;
      int n = t * 16 + ln;
      float v = acc[t][i] - (m == n ? 1e8f : 0.0f);
      adjF[m * 64 + n] = mpnn_lrelu(v);
    }
  __syncthreads();

  // ---- row softmax + eye ----
  if (tid < 64) {
    const float* row = &adjF[tid * 64];
    float mx = -1e30f;
    for (int j = 0; j < 64; ++j) mx = fmaxf(mx, row[j]);
    float s = 0.0f;
    for (int j = 0; j < 64; ++j) s += expf(row[j] - mx);
    float inv = 1.0f / s;
    for (int j = 0; j < 64; ++j)
      adjB[tid * 64 + j] =
          (__bf16)(expf(row[j] - mx) * inv + (j == tid ? 1.0f : 0.0f));
  }
  __syncthreads();

  // ---- build nfT[f][m] = nfL[m][f]  (adjF is dead now) ----
  for (int i = tid; i < 64 * 128; i += 128) {
    int m = i >> 7;
    int f = i & 127;
    nfT[f * 64 + m] = nfL[i];
  }
  __syncthreads();

  // ---- h = adj @ nf  (M=64,N=128,K=64) ----
  v8f hacc[8];
#pragma unroll
  for (int t = 0; t < 8; ++t) hacc[t] = mpnn_zero8();
#pragma unroll
  for (int ks = 0; ks < 2; ++ks) {
    v16bf af = mpnn_afrag(adjB + (size_t)(wv * 16 + ln) * 64, ks * 32, g);
#pragma unroll
    for (int t = 0; t < 8; ++t) {
      v16bf bf_ = mpnn_bfrag(nfT + (size_t)(t * 16 + ln) * 64, ks * 32, g);
      hacc[t] = mpnn_wmma(af, bf_, hacc[t]);
    }
  }
#pragma unroll
  for (int t = 0; t < 8; ++t)
#pragma unroll
    for (int i = 0; i < 8; ++i)
      hB[(size_t)(wv * 16 + g * 8 + i) * 128 + t * 16 + ln] = (__bf16)hacc[t][i];
  __syncthreads();

  // ---- out = h @ theta^T + b  (M=64,N=64,K=128) ----
  v8f oacc[4];
#pragma unroll
  for (int t = 0; t < 4; ++t) oacc[t] = mpnn_zero8();
#pragma unroll
  for (int ks = 0; ks < 4; ++ks) {
    v16bf af = mpnn_afrag(hB + (size_t)(wv * 16 + ln) * 128, ks * 32, g);
#pragma unroll
    for (int t = 0; t < 4; ++t) {
      v16bf bf_ = mpnn_bfrag(thw + (size_t)(t * 16 + ln) * 128, ks * 32, g);
      oacc[t] = mpnn_wmma(af, bf_, oacc[t]);
    }
  }
#pragma unroll
  for (int t = 0; t < 4; ++t) {
    int o = t * 16 + ln;
    float bias = thb[o];
    float s1 = 0.0f, s2 = 0.0f;
#pragma unroll
    for (int i = 0; i < 8; ++i) {
      int m = wv * 16 + g * 8 + i;
      float v = oacc[t][i] + bias;
      G[((size_t)bt * 64 + m) * 64 + o] = v;
      s1 += v;
      s2 += v * v;
    }
    atomicAdd(&sst[o], s1);
    atomicAdd(&sst[64 + o], s2);
  }
  __syncthreads();
  atomicAdd(&gstats[tid], sst[tid]);
}

// ---------------------------------------------------------------------------
// Final: BN + leaky_relu + mean over pairs of time steps -> [16,36,64,64] f32
// ---------------------------------------------------------------------------
__global__ void mpnn_final_out(const float* __restrict__ G,
                               const float* __restrict__ ss,
                               float* __restrict__ out) {
  int idx = blockIdx.x * 256 + threadIdx.x;  // 16*36*64*64 = 2359296
  if (idx >= 16 * 36 * 64 * 64) return;
  int o = idx & 63;
  int n = (idx >> 6) & 63;
  int wb = idx >> 12;
  int w = wb % 36;
  int b = wb / 36;
  float sc = ss[o], sh = ss[64 + o];
  size_t bt0 = (size_t)b * 72 + 2 * w;
  float v0 = mpnn_lrelu(G[(bt0 * 64 + n) * 64 + o] * sc + sh);
  float v1 = mpnn_lrelu(G[((bt0 + 1) * 64 + n) * 64 + o] * sc + sh);
  out[idx] = 0.5f * (v0 + v1);
}

// ---------------------------------------------------------------------------
// Host launcher
// ---------------------------------------------------------------------------
extern "C" void kernel_launch(void* const* d_in, const int* in_sizes, int n_in,
                              void* d_out, int out_size, void* d_ws, size_t ws_size,
                              hipStream_t stream) {
  const float* x  = (const float*)d_in[0];
  const float* w1 = (const float*)d_in[1];
  const float* g1 = (const float*)d_in[2];
  const float* b1 = (const float*)d_in[3];
  const float* w2 = (const float*)d_in[4];
  const float* g2 = (const float*)d_in[5];
  const float* b2 = (const float*)d_in[6];
  const float* w3 = (const float*)d_in[7];
  const float* g3 = (const float*)d_in[8];
  const float* b3 = (const float*)d_in[9];
  const float* mw = (const float*)d_in[10];
  const float* mb = (const float*)d_in[11];
  const float* tw = (const float*)d_in[12];
  const float* tb = (const float*)d_in[13];
  const float* bg = (const float*)d_in[14];
  const float* bb = (const float*)d_in[15];
  float* out = (float*)d_out;

  char* ws = (char*)d_ws;
  size_t off = 0;
  auto take = [&](size_t bytes) {
    size_t o = off;
    off += (bytes + 255) & ~(size_t)255;
    return o;
  };
  float*  ST  = (float*)(ws + take(8192));        // bn stats (zeroed each call)
  float*  SS  = (float*)(ws + take(8192));        // scale/shift
  __bf16* WB1 = (__bf16*)(ws + take(2ull * 256 * 192));
  __bf16* WB2 = (__bf16*)(ws + take(2ull * 128 * 768));
  __bf16* WB3 = (__bf16*)(ws + take(2ull * 256 * 384));
  __bf16* MWB = (__bf16*)(ws + take(2ull * 128 * 128));
  __bf16* TWB = (__bf16*)(ws + take(2ull * 64 * 128));
  char*   X0r = ws + take(2ull * 1024 * 64 * 256);   // X0, later spa
  char*   BIG = ws + take(2ull * 1024 * 256 * 256);  // conv1/2/3 outputs
  char*   P1r = ws + take(2ull * 1024 * 256 * 129);  // pool1, later nf + G
  __bf16* P2  = (__bf16*)(ws + take(2ull * 1024 * 128 * 66));
  __bf16* P3  = (__bf16*)(ws + take(2ull * 1024 * 256 * 36));

  __bf16* X0  = (__bf16*)X0r;
  __bf16* SPA = (__bf16*)X0r;        // reuse (X0 dead after conv1)
  __bf16* CBF = (__bf16*)BIG;        // reused conv output buffer
  __bf16* P1  = (__bf16*)P1r;
  __bf16* NF  = (__bf16*)P1r;                           // reuse (P1 dead after conv2)
  float*  G   = (float*)(P1r + 2ull * 1152 * 64 * 128); // next to NF inside P1 region

  float* ST1 = ST, * ST2 = ST + 512, * ST3 = ST + 768, * STG = ST + 1280;
  float* SS1 = SS, * SS2 = SS + 512, * SS3 = SS + 768, * SSG = SS + 1280;

  // dynamic LDS sizes for the conv kernels: Xl (CIN*132) + Bt (4096) + sst (512)
  const size_t sh1 = (size_t)64 * 132 + 4096 + 512;
  const size_t sh2 = (size_t)256 * 132 + 4096 + 512;
  const size_t sh3 = (size_t)128 * 132 + 4096 + 512;

  // 0) zero stats
  mpnn_zero_f32<<<(1408 + 255) / 256, 256, 0, stream>>>(ST, 1408);

  // 1) weight conversion to bf16 (flat [Cout][Cin*3] matches WMMA K order)
  mpnn_cvt_bf16<<<(49152 + 255) / 256, 256, 0, stream>>>(w1, WB1, 49152);
  mpnn_cvt_bf16<<<(98304 + 255) / 256, 256, 0, stream>>>(w2, WB2, 98304);
  mpnn_cvt_bf16<<<(98304 + 255) / 256, 256, 0, stream>>>(w3, WB3, 98304);
  mpnn_cvt_bf16<<<(16384 + 255) / 256, 256, 0, stream>>>(mw, MWB, 16384);
  mpnn_cvt_bf16<<<(8192 + 255) / 256, 256, 0, stream>>>(tw, TWB, 8192);

  // 2) pack input
  mpnn_pack_x<<<16777216 / 256, 256, 0, stream>>>(x, X0);

  // 3) block 1: conv(64->256,pad1,L256) + BN + relu + pool -> [1024,256,129]
  mpnn_conv_bn<64, 256, 256, 256, 1>
      <<<dim3(1024, 4, 4), 128, sh1, stream>>>(X0, WB1, CBF, ST1);
  mpnn_finalize_bn<<<1, 256, 0, stream>>>(ST1, g1, b1, 1024.0f * 256.0f, 256, SS1);
  mpnn_bn_relu_pool<256, 256, 129>
      <<<(1024 * 256 * 129 + 255) / 256, 256, 0, stream>>>(CBF, SS1, P1);

  // 4) block 2: conv(256->128,pad2,L129->131) + BN + relu + pool -> [1024,128,66]
  mpnn_conv_bn<256, 128, 129, 131, 2>
      <<<dim3(1024, 2, 3), 128, sh2, stream>>>(P1, WB2, CBF, ST2);
  mpnn_finalize_bn<<<1, 256, 0, stream>>>(ST2, g2, b2, 1024.0f * 131.0f, 128, SS2);
  mpnn_bn_relu_pool<128, 131, 66>
      <<<(1024 * 128 * 66 + 255) / 256, 256, 0, stream>>>(CBF, SS2, P2);

  // 5) block 3: conv(128->256,pad3,L66->70) + BN + relu + pool -> [1024,256,36]
  mpnn_conv_bn<128, 256, 66, 70, 3>
      <<<dim3(1024, 4, 2), 128, sh3, stream>>>(P2, WB3, CBF, ST3);
  mpnn_finalize_bn<<<1, 256, 0, stream>>>(ST3, g3, b3, 1024.0f * 70.0f, 256, SS3);
  mpnn_bn_relu_pool<256, 70, 36>
      <<<(1024 * 256 * 36 + 255) / 256, 256, 0, stream>>>(CBF, SS3, P3);

  // 6) reshape to spatial batches
  mpnn_pack_spa<<<(1152 * 64 * 128 + 255) / 256, 256, 0, stream>>>(P3, SPA);

  // 7) node features
  mpnn_nf_gemm<<<1152, 256, 0, stream>>>(SPA, MWB, mb, NF);

  // 8) graph construction + MPNN (+ BN stats)
  mpnn_graph<<<1152, 128, 0, stream>>>(NF, TWB, tb, G, STG);
  mpnn_finalize_bn<<<1, 256, 0, stream>>>(STG, bg, bb, 1152.0f * 64.0f, 64, SSG);

  // 9) BN + leaky_relu + window pooling -> output
  mpnn_final_out<<<(16 * 36 * 64 * 64 + 255) / 256, 256, 0, stream>>>(G, SSG, out);

  (void)in_sizes; (void)n_in; (void)out_size; (void)ws_size;
}
